// FlaxAttention_10634339025458
// MI455X (gfx1250) — matrix operand order repaired and checked by reading, not compiled, not run on hardware
//
#include <hip/hip_runtime.h>

// ---------------- types ----------------
typedef __bf16 bf16_t;
typedef __attribute__((ext_vector_type(16))) __bf16 v16bf;
typedef __attribute__((ext_vector_type(8)))  __bf16 v8bf;
typedef __attribute__((ext_vector_type(8)))  float  v8f;

#define D_MODEL 2048
#define SEQ     2048
#define NHEADS  16
#define HDIM    128
#define BATCH   2
#define MROWS   (BATCH * SEQ)   // 4096

static __device__ inline v8f zero_v8f() {
    v8f z;
#pragma unroll
    for (int i = 0; i < 8; ++i) z[i] = 0.0f;
    return z;
}

union BF16x16 { v16bf v; v8bf p[2]; };

// ---------------- fp32 -> bf16 conversion ----------------
__global__ __launch_bounds__(256)
void f32_to_bf16_kernel(const float* __restrict__ in, bf16_t* __restrict__ out, int n4) {
    int i = (blockIdx.x * 256 + threadIdx.x);
    if (i < n4) {
        const float4 f = ((const float4*)in)[i];
        bf16_t* o = out + (size_t)i * 4;
        o[0] = (bf16_t)f.x; o[1] = (bf16_t)f.y; o[2] = (bf16_t)f.z; o[3] = (bf16_t)f.w;
    }
}

// ---------------- bf16 WMMA GEMM: C[M,N] = A[M,K] * B[K,N] ----------------
// N, K compile-time so the epilogue folds into immediate store offsets.
// Block tile 128x128, BK=32, 256 threads = 8 waves (2x4), wave tile 64x32.
// Register double-buffered: next tile's global loads overlap current WMMAs.
template <int N, int K>
__global__ __launch_bounds__(256)
void gemm_bf16_kernel(const bf16_t* __restrict__ A, const bf16_t* __restrict__ Bm,
                      float* __restrict__ C) {
    __shared__ __align__(16) bf16_t As[128][40];   // row-major A tile (+pad)
    __shared__ __align__(16) bf16_t BsT[128][40];  // B tile transposed: BsT[n][k]

    const int tid  = threadIdx.x;
    const int lane = tid & 31;
    const int wv   = tid >> 5;
    const int hf   = lane >> 4;   // 16-lane half
    const int l15  = lane & 15;

    const int row0 = blockIdx.y * 128;
    const int col0 = blockIdx.x * 128;
    const int wm   = (wv >> 2) * 64;
    const int wn   = (wv & 3) * 32;

    // staging-thread coordinates
    const int sra = tid >> 1;             // A row 0..127
    const int sha = (tid & 1) * 16;       // A k-offset 0 or 16
    const int skb0 = (tid * 8) >> 7;      // B k for chunk 0
    const int snb0 = (tid * 8) & 127;     // B n for chunk 0
    const int skb1 = ((tid + 256) * 8) >> 7;
    const int snb1 = ((tid + 256) * 8) & 127;

    v8bf pa0, pa1, pb0, pb1;              // register double-buffer

    auto load_tile = [&](int k0) {
        const bf16_t* asrc = A + (size_t)(row0 + sra) * K + k0 + sha;
        pa0 = *(const v8bf*)asrc;
        pa1 = *(const v8bf*)(asrc + 8);
        pb0 = *(const v8bf*)(Bm + (size_t)(k0 + skb0) * N + col0 + snb0);
        pb1 = *(const v8bf*)(Bm + (size_t)(k0 + skb1) * N + col0 + snb1);
    };
    auto store_tile = [&]() {
        *(v8bf*)&As[sra][sha]     = pa0;
        *(v8bf*)&As[sra][sha + 8] = pa1;
#pragma unroll
        for (int j = 0; j < 8; ++j) BsT[snb0 + j][skb0] = pb0[j];
#pragma unroll
        for (int j = 0; j < 8; ++j) BsT[snb1 + j][skb1] = pb1[j];
    };

    v8f acc[4][2];
#pragma unroll
    for (int mt = 0; mt < 4; ++mt)
#pragma unroll
        for (int nt = 0; nt < 2; ++nt) acc[mt][nt] = zero_v8f();

    load_tile(0);
    store_tile();

    for (int k0 = 0; k0 < K; k0 += 32) {
        __syncthreads();
        const bool has_next = (k0 + 32 < K);
        if (has_next) load_tile(k0 + 32);                  // overlap with WMMAs below
        if (k0 + 64 < K) {                                 // prime caches one more ahead
            __builtin_prefetch(A + (size_t)(row0 + sra) * K + k0 + 64 + sha, 0, 1);
            __builtin_prefetch(Bm + (size_t)(k0 + 64 + skb0) * N + col0 + snb0, 0, 1);
        }

        // --- fragments from LDS (all loads issued before the WMMA chain) ---
        v16bf af[4];
#pragma unroll
        for (int mt = 0; mt < 4; ++mt) {
            const int rowA = wm + mt * 16 + l15;
            BF16x16 u;
            u.p[0] = *(const v8bf*)&As[rowA][hf * 8];        // K = hf*8 + 0..7
            u.p[1] = *(const v8bf*)&As[rowA][16 + hf * 8];   // K = 16 + hf*8 + 0..7
            af[mt] = u.v;
        }
        v16bf bfv[2];
#pragma unroll
        for (int nt = 0; nt < 2; ++nt) {
            const int nB = wn + nt * 16 + l15;
            BF16x16 u;
            u.p[0] = *(const v8bf*)&BsT[nB][hf * 8];         // K = hf*8 + 0..7
            u.p[1] = *(const v8bf*)&BsT[nB][16 + hf * 8];    // K = 16 + hf*8 + 0..7
            bfv[nt] = u.v;
        }
#pragma unroll
        for (int mt = 0; mt < 4; ++mt)
#pragma unroll
            for (int nt = 0; nt < 2; ++nt)
                acc[mt][nt] = __builtin_amdgcn_wmma_f32_16x16x32_bf16(
                    false, af[mt], false, bfv[nt], (short)0, acc[mt][nt], false, false);

        __syncthreads();
        if (has_next) store_tile();
    }

    // --- epilogue: constant offsets off one base pointer ---
    float* cbase = C + (size_t)(row0 + wm + hf * 8) * N + col0 + wn + l15;
#pragma unroll
    for (int mt = 0; mt < 4; ++mt)
#pragma unroll
        for (int nt = 0; nt < 2; ++nt)
#pragma unroll
            for (int r = 0; r < 8; ++r)
                cbase[(size_t)(mt * 16 + r) * N + nt * 16] = acc[mt][nt][r];
}

// ---------------- LayerNorm (scale only) on q,k + split to bf16 ----------------
// q additionally folds (1/sqrt(HDIM)) * log2(e) so attention runs in exp2 domain.
__global__ __launch_bounds__(256)
void ln_split_kernel(const float* __restrict__ qkv,
                     const float* __restrict__ q_scale,
                     const float* __restrict__ k_scale,
                     bf16_t* __restrict__ qb, bf16_t* __restrict__ kb,
                     bf16_t* __restrict__ vb) {
    const int row = blockIdx.x;       // 0..4095
    const int tid = threadIdx.x;
    __shared__ float red[4][8];

    const float* base = qkv + (size_t)row * (3 * D_MODEL);

    float vq[8], vk[8];
    float s1q = 0.f, s2q = 0.f, s1k = 0.f, s2k = 0.f;
#pragma unroll
    for (int i = 0; i < 8; ++i) {
        const int idx = tid + i * 256;
        const float q = base[idx];
        const float k = base[D_MODEL + idx];
        vq[i] = q; vk[i] = k;
        s1q += q; s2q += q * q; s1k += k; s2k += k * k;
    }
#pragma unroll
    for (int d = 1; d < 32; d <<= 1) {
        s1q += __shfl_xor(s1q, d, 32);
        s2q += __shfl_xor(s2q, d, 32);
        s1k += __shfl_xor(s1k, d, 32);
        s2k += __shfl_xor(s2k, d, 32);
    }
    const int wv = tid >> 5;
    if ((tid & 31) == 0) {
        red[0][wv] = s1q; red[1][wv] = s2q; red[2][wv] = s1k; red[3][wv] = s2k;
    }
    __syncthreads();
    float t1q = 0.f, t2q = 0.f, t1k = 0.f, t2k = 0.f;
#pragma unroll
    for (int i = 0; i < 8; ++i) {
        t1q += red[0][i]; t2q += red[1][i]; t1k += red[2][i]; t2k += red[3][i];
    }
    const float inv_n = 1.0f / (float)D_MODEL;
    const float mq  = t1q * inv_n;
    const float mk  = t1k * inv_n;
    const float ivq = rsqrtf(t2q * inv_n - mq * mq + 1e-6f);
    const float ivk = rsqrtf(t2k * inv_n - mk * mk + 1e-6f);
    const float qscl = 0.12751743f;   // (1/sqrt(HDIM)) * log2(e)

#pragma unroll
    for (int i = 0; i < 8; ++i) {
        const int idx = tid + i * 256;
        const size_t o = (size_t)row * D_MODEL + idx;
        qb[o] = (bf16_t)((vq[i] - mq) * ivq * q_scale[idx] * qscl);
        kb[o] = (bf16_t)((vk[i] - mk) * ivk * k_scale[idx]);
        vb[o] = (bf16_t)(base[2 * D_MODEL + idx]);
    }
}

// ---------------- causal flash attention (bf16 WMMA, online softmax) ----------------
// grid: (SEQ/128, NHEADS, BATCH), 256 threads = 8 waves.
// Wave w owns query rows q0 = blk*128 + w*16 .. +16.  K/V streamed in 32-key tiles,
// register double-buffered.  Softmax in exp2 domain; row sums via P @ ones WMMA;
// all LDS fragment loads issued before their WMMA chains.
__global__ __launch_bounds__(256)
void flash_attn_kernel(const bf16_t* __restrict__ Q, const bf16_t* __restrict__ Kmat,
                       const bf16_t* __restrict__ Vmat, bf16_t* __restrict__ Ctx) {
    const int qblk = blockIdx.x;
    const int h    = blockIdx.y;
    const int b    = blockIdx.z;
    const int tid  = threadIdx.x;
    const int lane = tid & 31;
    const int wv   = tid >> 5;
    const int hf   = lane >> 4;
    const int l15  = lane & 15;

    __shared__ __align__(16) bf16_t Ks[32][HDIM + 8];   // [key][d]
    __shared__ __align__(16) bf16_t VsT[HDIM][40];      // [d][key]
    __shared__ __align__(16) bf16_t Ps[8][16][40];      // per-wave P bounce

    const int q0 = qblk * 128 + wv * 16;
    const size_t base_bh = ((size_t)b * SEQ) * D_MODEL + (size_t)h * HDIM;

    // Q fragments (A-matrix 16x32, held for the whole kernel; K-dim = HDIM/32 = 4)
    v16bf qf[4];
    {
        const size_t qrow = base_bh + (size_t)(q0 + l15) * D_MODEL;
#pragma unroll
        for (int ks = 0; ks < 4; ++ks) {
            BF16x16 u;
            u.p[0] = *(const v8bf*)(Q + qrow + ks * 32 + hf * 8);
            u.p[1] = *(const v8bf*)(Q + qrow + ks * 32 + 16 + hf * 8);
            qf[ks] = u.v;
        }
    }

    // constant all-ones B fragment: rowsum(P) = P @ ones
    v16bf ones_v;
#pragma unroll
    for (int j = 0; j < 16; ++j) ones_v[j] = (bf16_t)1.0f;

    v8f O[8];
#pragma unroll
    for (int i = 0; i < 8; ++i) O[i] = zero_v8f();
    v8f Lacc = zero_v8f();               // running row sums (replicated per half)
    float m_i[8];
#pragma unroll
    for (int r = 0; r < 8; ++r) m_i[r] = -1e30f;

    const int kmax = qblk * 128 + 128;   // causal upper bound for this block

    // staging coordinates + register double-buffer for K/V tiles
    const int skey = (tid * 16) >> 7;
    const int sd   = (tid * 16) & 127;
    v8bf kp0, kp1, vp0, vp1;

    auto load_kv = [&](int kb) {
        const bf16_t* ksrc = Kmat + base_bh + (size_t)(kb + skey) * D_MODEL + sd;
        kp0 = *(const v8bf*)ksrc;
        kp1 = *(const v8bf*)(ksrc + 8);
        const bf16_t* vsrc = Vmat + base_bh + (size_t)(kb + skey) * D_MODEL + sd;
        vp0 = *(const v8bf*)vsrc;
        vp1 = *(const v8bf*)(vsrc + 8);
    };
    auto store_kv = [&]() {
        *(v8bf*)&Ks[skey][sd]     = kp0;
        *(v8bf*)&Ks[skey][sd + 8] = kp1;
#pragma unroll
        for (int j = 0; j < 8; ++j) {
            VsT[sd + j][skey]     = vp0[j];
            VsT[sd + 8 + j][skey] = vp1[j];
        }
    };

    load_kv(0);
    store_kv();

    for (int kb = 0; kb < kmax; kb += 32) {
        __syncthreads();
        const bool has_next = (kb + 32 < kmax);
        if (has_next) load_kv(kb + 32);                    // overlap with compute
        if (kb + 64 < kmax)
            __builtin_prefetch(Kmat + base_bh + (size_t)(kb + 64 + skey) * D_MODEL + sd, 0, 1);

        // wave-uniform skip: tiles entirely above the diagonal contribute nothing
        if (kb <= q0 + 15) {
            // --- scores: S = Q @ K^T; issue ALL 8 K-fragment loads, then WMMA chain ---
            BF16x16 kf[8];
#pragma unroll
            for (int ks = 0; ks < 4; ++ks) {
                kf[2 * ks].p[0]     = *(const v8bf*)&Ks[l15][ks * 32 + hf * 8];
                kf[2 * ks].p[1]     = *(const v8bf*)&Ks[l15][ks * 32 + 16 + hf * 8];
                kf[2 * ks + 1].p[0] = *(const v8bf*)&Ks[16 + l15][ks * 32 + hf * 8];
                kf[2 * ks + 1].p[1] = *(const v8bf*)&Ks[16 + l15][ks * 32 + 16 + hf * 8];
            }
            v8f S0 = zero_v8f(), S1 = zero_v8f();
#pragma unroll
            for (int ks = 0; ks < 4; ++ks) {
                S0 = __builtin_amdgcn_wmma_f32_16x16x32_bf16(false, qf[ks], false, kf[2 * ks].v,
                                                             (short)0, S0, false, false);
                S1 = __builtin_amdgcn_wmma_f32_16x16x32_bf16(false, qf[ks], false, kf[2 * ks + 1].v,
                                                             (short)0, S1, false, false);
            }

            // --- online softmax (exp2 domain); mask compares only on diagonal tiles ---
            auto softmax_pass = [&](bool domask) {
#pragma unroll
                for (int r = 0; r < 8; ++r) {
                    const int row = q0 + hf * 8 + r;
                    float s0 = S0[r], s1 = S1[r];
                    if (domask) {
                        const int c0 = kb + l15;
                        if (c0 > row)      s0 = -1e30f;
                        if (c0 + 16 > row) s1 = -1e30f;
                    }
                    float mx = fmaxf(s0, s1);
#pragma unroll
                    for (int dd = 1; dd < 16; dd <<= 1)
                        mx = fmaxf(mx, __shfl_xor(mx, dd, 32));
                    const float m_new = fmaxf(m_i[r], mx);
                    const float alpha = exp2f(m_i[r] - m_new);
                    m_i[r] = m_new;
                    const float p0 = exp2f(s0 - m_new);
                    const float p1 = exp2f(s1 - m_new);
                    Lacc[r] *= alpha;
#pragma unroll
                    for (int t = 0; t < 8; ++t) O[t][r] *= alpha;
                    Ps[wv][hf * 8 + r][l15]      = (bf16_t)p0;
                    Ps[wv][hf * 8 + r][16 + l15] = (bf16_t)p1;
                }
            };
            if (kb + 31 > q0) softmax_pass(true);
            else              softmax_pass(false);

            // --- PV phase: issue all 8 V-fragment loads, then the P bounce read ---
            BF16x16 vf[8];
#pragma unroll
            for (int nt = 0; nt < 8; ++nt) {
                vf[nt].p[0] = *(const v8bf*)&VsT[nt * 16 + l15][hf * 8];
                vf[nt].p[1] = *(const v8bf*)&VsT[nt * 16 + l15][16 + hf * 8];
            }
            BF16x16 pf;
            pf.p[0] = *(const v8bf*)&Ps[wv][l15][hf * 8];
            pf.p[1] = *(const v8bf*)&Ps[wv][l15][16 + hf * 8];

            // row sums: Lacc += P @ ones  (replaces a 32-op shuffle tree)
            Lacc = __builtin_amdgcn_wmma_f32_16x16x32_bf16(false, pf.v, false, ones_v,
                                                           (short)0, Lacc, false, false);
#pragma unroll
            for (int nt = 0; nt < 8; ++nt)
                O[nt] = __builtin_amdgcn_wmma_f32_16x16x32_bf16(false, pf.v, false, vf[nt].v,
                                                                (short)0, O[nt], false, false);
        }

        __syncthreads();
        if (has_next) store_kv();
    }

    // --- epilogue: normalize and store ctx (bf16) ---
#pragma unroll
    for (int r = 0; r < 8; ++r) {
        const float l = Lacc[r];
        const float inv = (l > 0.f) ? (1.0f / l) : 0.f;
        const size_t orow = base_bh + (size_t)(q0 + hf * 8 + r) * D_MODEL;
#pragma unroll
        for (int nt = 0; nt < 8; ++nt)
            Ctx[orow + nt * 16 + l15] = (bf16_t)(O[nt][r] * inv);
    }
}

// ---------------- launch ----------------
extern "C" void kernel_launch(void* const* d_in, const int* in_sizes, int n_in,
                              void* d_out, int out_size, void* d_ws, size_t ws_size,
                              hipStream_t stream) {
    const float* x     = (const float*)d_in[0];
    const float* Wqkv  = (const float*)d_in[1];
    const float* q_ln  = (const float*)d_in[2];
    const float* k_ln  = (const float*)d_in[3];
    const float* out_W = (const float*)d_in[4];
    float* out = (float*)d_out;

    char* w = (char*)d_ws;
    bf16_t* xb    = (bf16_t*)w; w += (size_t)MROWS * D_MODEL * 2;
    bf16_t* wqkvb = (bf16_t*)w; w += (size_t)D_MODEL * 3 * D_MODEL * 2;
    bf16_t* outwb = (bf16_t*)w; w += (size_t)D_MODEL * D_MODEL * 2;
    float*  qkv   = (float*) w; w += (size_t)MROWS * 3 * D_MODEL * 4;
    bf16_t* qb    = (bf16_t*)w; w += (size_t)MROWS * D_MODEL * 2;
    bf16_t* kbuf  = (bf16_t*)w; w += (size_t)MROWS * D_MODEL * 2;
    bf16_t* vbuf  = (bf16_t*)w; w += (size_t)MROWS * D_MODEL * 2;
    bf16_t* ctxb  = (bf16_t*)w; w += (size_t)MROWS * D_MODEL * 2;

    // 1) fp32 -> bf16 conversions
    {
        const int n1 = MROWS * D_MODEL / 4;
        f32_to_bf16_kernel<<<(n1 + 255) / 256, 256, 0, stream>>>(x, xb, n1);
        const int n2 = D_MODEL * 3 * D_MODEL / 4;
        f32_to_bf16_kernel<<<(n2 + 255) / 256, 256, 0, stream>>>(Wqkv, wqkvb, n2);
        const int n3 = D_MODEL * D_MODEL / 4;
        f32_to_bf16_kernel<<<(n3 + 255) / 256, 256, 0, stream>>>(out_W, outwb, n3);
    }
    // 2) qkv = x @ Wqkv   [4096 x 6144]
    {
        dim3 g(3 * D_MODEL / 128, MROWS / 128);
        gemm_bf16_kernel<3 * D_MODEL, D_MODEL><<<g, 256, 0, stream>>>(xb, wqkvb, qkv);
    }
    // 3) LayerNorm(q), LayerNorm(k), split to bf16
    ln_split_kernel<<<MROWS, 256, 0, stream>>>(qkv, q_ln, k_ln, qb, kbuf, vbuf);
    // 4) causal flash attention -> ctx
    {
        dim3 g(SEQ / 128, NHEADS, BATCH);
        flash_attn_kernel<<<g, 256, 0, stream>>>(qb, kbuf, vbuf, ctxb);
    }
    // 5) out = ctx @ out_W   [4096 x 2048]
    {
        dim3 g(D_MODEL / 128, MROWS / 128);
        gemm_bf16_kernel<D_MODEL, D_MODEL><<<g, 256, 0, stream>>>(ctxb, outwb, out);
    }
}
